// MyBertAttnBPWordPiece_43267500540798
// MI455X (gfx1250) — matrix operand order repaired and checked by reading, not compiled, not run on hardware
//
#include <hip/hip_runtime.h>

typedef __attribute__((ext_vector_type(16))) __bf16 v16bf;
typedef __attribute__((ext_vector_type(8)))  __bf16 v8bf;
typedef __attribute__((ext_vector_type(8)))  float  v8f;

#define HDIM   768
#define BATCH  256
#define SEQ    256
#define NW     51
#define INTENT 26
#define SLOT   122
#define NPAIR  (BATCH*NW*2)   /* 26112 pair rows  */
#define NWORD  (BATCH*NW)     /* 13056 word rows  */

#if __has_builtin(__builtin_amdgcn_tanhf)
#define TANHF(x) __builtin_amdgcn_tanhf(x)
#else
#define TANHF(x) tanhf(x)
#endif

__device__ __forceinline__ unsigned short f2bf(float f) {
    unsigned int u = __float_as_uint(f);
    unsigned int r = u + 0x7FFFu + ((u >> 16) & 1u);   // round-to-nearest-even
    return (unsigned short)(r >> 16);
}
__device__ __forceinline__ float bf2f(unsigned short s) {
    return __uint_as_float(((unsigned int)s) << 16);
}

// gfx1250 async global->LDS copy, 16B per lane (tracked by ASYNCcnt).
// Generic LDS pointer's low 32 bits are the LDS byte address.
__device__ __forceinline__ void async_b128(unsigned short* lds_dst,
                                           const unsigned short* gsrc) {
    unsigned off = (unsigned)(uintptr_t)lds_dst;
    asm volatile("global_load_async_to_lds_b128 %0, %1, off"
                 :: "v"(off), "v"(gsrc) : "memory");
}
#if __has_builtin(__builtin_amdgcn_s_wait_asynccnt)
__device__ __forceinline__ void wait_async_le2() { __builtin_amdgcn_s_wait_asynccnt((short)2); }
__device__ __forceinline__ void wait_async_0()  { __builtin_amdgcn_s_wait_asynccnt((short)0); }
#else
__device__ __forceinline__ void wait_async_le2() {
    asm volatile("s_wait_asynccnt 2" ::: "memory");
}
__device__ __forceinline__ void wait_async_0() {
    asm volatile("s_wait_asynccnt 0" ::: "memory");
}
#endif

// build v16bf fragment from two 16B LDS chunks
__device__ __forceinline__ v16bf frag16(const unsigned short* p0,
                                        const unsigned short* p1) {
    v8bf lo = *(const v8bf*)p0;
    v8bf hi = *(const v8bf*)p1;
    return __builtin_shufflevector(lo, hi,
        0,1,2,3,4,5,6,7,8,9,10,11,12,13,14,15);
}

// ---------------------------------------------------------------------------
// bf16-pack attn_W [768x768] and slot_W padded to [128x768] (rows 122..127 = 0)
// ---------------------------------------------------------------------------
__global__ void pack_weights(const float* __restrict__ attn_W,
                             const float* __restrict__ slot_W,
                             unsigned short* __restrict__ Wb,
                             unsigned short* __restrict__ Sb) {
    const int idx = blockIdx.x * blockDim.x + threadIdx.x;
    if (idx < HDIM * HDIM) Wb[idx] = f2bf(attn_W[idx]);
    if (idx < 128 * HDIM) {
        const int o = idx / HDIM;
        Sb[idx] = (o < SLOT) ? f2bf(slot_W[idx]) : (unsigned short)0;
    }
}

// ---------------------------------------------------------------------------
// Fully fused: gather pair rows -> bf16 LDS tile -> energy GEMM (WMMA) ->
// tanh/v-dot scores -> softmax(k=2) -> pooled word tile in LDS ->
// slot GEMM (WMMA) -> res_sf.  One block = 16 words = 32 pair rows.
// B strips stream through LDS via async-to-LDS double buffering.
// ---------------------------------------------------------------------------
__global__ __launch_bounds__(256)
void fused_attn_slot(const float* __restrict__ res_all,
                     const int* __restrict__ starts,
                     const int* __restrict__ lens,
                     const unsigned short* __restrict__ Wb,
                     const unsigned short* __restrict__ Sb,
                     const float* __restrict__ attn_b,
                     const float* __restrict__ vw,
                     const float* __restrict__ slot_b,
                     float* __restrict__ out_sf) {
    __shared__ unsigned short ldsP[32 * HDIM];   // 48 KB pair tile (bf16)
    __shared__ unsigned short ldsW[16 * HDIM];   // 24 KB word tile (bf16)
    __shared__ unsigned short ldsB[8 * 1024];    // 16 KB: per-wave 2x512 B strip
    __shared__ float sscore[32];
    __shared__ float scoef[16][2];

    const int tid    = threadIdx.x;
    const int wave   = tid >> 5;
    const int lane   = tid & 31;
    const int laneHi = lane >> 4;        // fragment lane-half (ISA layout)
    const int l15    = lane & 15;

    if (tid < 32) sscore[tid] = 0.0f;

    // ---- phase 1: gather 32 pair rows (start / start+len quirk), f32->bf16
    {
        const int lr   = tid >> 3;                 // local pair row 0..31
        const int part = tid & 7;
        const int prow = blockIdx.x * 32 + lr;
        const int b    = prow / (NW * 2);
        const int r2   = prow - b * (NW * 2);
        const int n    = r2 >> 1, kk = r2 & 1;
        const int st   = starts[b * NW + n];
        const int ln   = lens[b * NW + n];
        int src = st + (kk ? ln : 0);
        src = src < 0 ? 0 : (src > SEQ - 1 ? SEQ - 1 : src);
        const float4* sp = (const float4*)(res_all + ((size_t)b * SEQ + src) * HDIM) + part * 24;
        uint2* dp = (uint2*)&ldsP[lr * HDIM] + part * 24;
        #pragma unroll 4
        for (int i = 0; i < 24; ++i) {
            float4 x = sp[i];
            uint2 o;
            o.x = (unsigned)f2bf(x.x) | ((unsigned)f2bf(x.y) << 16);
            o.y = (unsigned)f2bf(x.z) | ((unsigned)f2bf(x.w) << 16);
            dp[i] = o;
        }
    }
    __syncthreads();

    // ---- phase 2: energy GEMM + tanh + v-dot (two 16-row subtiles per wave)
    float sacc0[8], sacc1[8];
    #pragma unroll
    for (int r = 0; r < 8; ++r) { sacc0[r] = 0.0f; sacc1[r] = 0.0f; }

    const int colA = lane >> 2;          // async copy: col 0..7 (+8 on 2nd issue)
    const int subk = lane & 3;           // 16B sub-chunk within 64B K-run
    unsigned short* myB = &ldsB[wave * 1024];
    unsigned short* myB0 = myB + colA * 32 + subk * 8;

    for (int c = 0; c < 6; ++c) {
        const int n0 = c * 128 + wave * 16;
        const float bias = attn_b[n0 + l15];            // hoisted epilogue scalars
        const float vn   = vw[n0 + l15];
        const unsigned short* g0 = Wb + (size_t)(n0 + colA)     * HDIM + subk * 8;
        const unsigned short* g1 = Wb + (size_t)(n0 + colA + 8) * HDIM + subk * 8;
        // prologue: kt=0 strip into buffer 0
        async_b128(myB0,       g0);
        async_b128(myB0 + 256, g1);

        v8f d0 = {}; v8f d1 = {};
        #pragma unroll
        for (int kt = 0; kt < 24; ++kt) {
            // A frags are independent of async traffic: issue before the wait
            const int ka = kt * 32 + laneHi * 8;
            v16bf a0 = frag16(&ldsP[l15 * HDIM + ka],        &ldsP[l15 * HDIM + ka + 16]);
            v16bf a1 = frag16(&ldsP[(l15 + 16) * HDIM + ka], &ldsP[(l15 + 16) * HDIM + ka + 16]);
            if (kt < 23) {   // pipeline next strip into the other buffer
                unsigned short* dn = myB0 + ((kt + 1) & 1) * 512;
                async_b128(dn,       g0 + (kt + 1) * 32);
                async_b128(dn + 256, g1 + (kt + 1) * 32);
                wait_async_le2();
            } else {
                wait_async_0();
            }
            const unsigned short* bp = myB + (kt & 1) * 512 + l15 * 32 + laneHi * 8;
            v16bf bv = frag16(bp, bp + 16);
            d0 = __builtin_amdgcn_wmma_f32_16x16x32_bf16(false, a0, false, bv, (short)0, d0, false, false);
            d1 = __builtin_amdgcn_wmma_f32_16x16x32_bf16(false, a1, false, bv, (short)0, d1, false, false);
        }
        #pragma unroll
        for (int r = 0; r < 8; ++r) {
            sacc0[r] += TANHF(d0[r] + bias) * vn;
            sacc1[r] += TANHF(d1[r] + bias) * vn;
        }
    }
    // reduce across the 16 lanes of each half (xor stays within half)
    #pragma unroll
    for (int r = 0; r < 8; ++r) {
        float s0 = sacc0[r], s1 = sacc1[r];
        s0 += __shfl_xor(s0, 1, 32); s0 += __shfl_xor(s0, 2, 32);
        s0 += __shfl_xor(s0, 4, 32); s0 += __shfl_xor(s0, 8, 32);
        s1 += __shfl_xor(s1, 1, 32); s1 += __shfl_xor(s1, 2, 32);
        s1 += __shfl_xor(s1, 4, 32); s1 += __shfl_xor(s1, 8, 32);
        sacc0[r] = s0; sacc1[r] = s1;
    }
    if (l15 == 0) {
        #pragma unroll
        for (int r = 0; r < 8; ++r) {
            atomicAdd(&sscore[laneHi * 8 + r],      sacc0[r]);
            atomicAdd(&sscore[16 + laneHi * 8 + r], sacc1[r]);
        }
    }
    __syncthreads();

    // ---- phase 3: per-word softmax coefficients
    if (tid < 16) {
        const int wg = blockIdx.x * 16 + tid;
        const int b  = wg / NW, n = wg - b * NW;
        const float s0 = sscore[2 * tid], s1 = sscore[2 * tid + 1];
        const float m  = fmaxf(s0, s1);
        const float e0 = __expf(s0 - m), e1 = __expf(s1 - m);
        const float inv = 1.0f / (e0 + e1);
        const int ln = lens[b * NW + n];
        scoef[tid][0] = (ln == 1) ? 1.0f : 1.0f + e0 * inv;   // e0 + w0*e0 + w1*e1
        scoef[tid][1] = (ln == 1) ? 0.0f : e1 * inv;
    }
    __syncthreads();

    // ---- phase 4: pooled word tile into LDS (16 x 768 bf16)
    for (int idx = tid; idx < 16 * HDIM; idx += 256) {
        const int j = idx / HDIM, h = idx - j * HDIM;
        const float val = scoef[j][0] * bf2f(ldsP[(2 * j) * HDIM + h]) +
                          scoef[j][1] * bf2f(ldsP[(2 * j + 1) * HDIM + h]);
        ldsW[idx] = f2bf(val);
    }
    __syncthreads();

    // ---- phase 5: slot GEMM, 8 waves cover N=128 (>=122), async B pipeline
    {
        const int n0 = wave * 16;
        const int ns = n0 + l15;
        const unsigned short* g0 = Sb + (size_t)(n0 + colA)     * HDIM + subk * 8;
        const unsigned short* g1 = Sb + (size_t)(n0 + colA + 8) * HDIM + subk * 8;
        async_b128(myB0,       g0);
        async_b128(myB0 + 256, g1);

        v8f d = {};
        #pragma unroll
        for (int kt = 0; kt < 24; ++kt) {
            const int ka = kt * 32 + laneHi * 8;
            v16bf av = frag16(&ldsW[l15 * HDIM + ka], &ldsW[l15 * HDIM + ka + 16]);
            if (kt < 23) {
                unsigned short* dn = myB0 + ((kt + 1) & 1) * 512;
                async_b128(dn,       g0 + (kt + 1) * 32);
                async_b128(dn + 256, g1 + (kt + 1) * 32);
                wait_async_le2();
            } else {
                wait_async_0();
            }
            const unsigned short* bp = myB + (kt & 1) * 512 + l15 * 32 + laneHi * 8;
            v16bf bv = frag16(bp, bp + 16);
            d = __builtin_amdgcn_wmma_f32_16x16x32_bf16(false, av, false, bv, (short)0, d, false, false);
        }
        if (ns < SLOT) {
            const float bias = slot_b[ns];
            #pragma unroll
            for (int r = 0; r < 8; ++r) {
                const size_t row = (size_t)blockIdx.x * 16 + r + laneHi * 8;
                out_sf[row * SLOT + ns] = d[r] + bias;
            }
        }
    }
}

// ---------------------------------------------------------------------------
// res_id = res @ id_Wt + id_b   (10 MFLOP -> exact fp32 VALU)
// ---------------------------------------------------------------------------
__global__ void intent_fc(const float* __restrict__ res,
                          const float* __restrict__ id_W,
                          const float* __restrict__ id_b,
                          float* __restrict__ out) {
    const int idx = blockIdx.x * blockDim.x + threadIdx.x;
    if (idx >= BATCH * INTENT) return;
    const int b = idx / INTENT, i = idx - b * INTENT;
    const float* x = res  + (size_t)b * HDIM;
    const float* w = id_W + (size_t)i * HDIM;
    float acc = 0.0f;
    for (int h = 0; h < HDIM; ++h) acc = fmaf(x[h], w[h], acc);
    out[idx] = acc + id_b[i];
}

// ---------------------------------------------------------------------------
extern "C" void kernel_launch(void* const* d_in, const int* in_sizes, int n_in,
                              void* d_out, int out_size, void* d_ws, size_t ws_size,
                              hipStream_t stream) {
    const float* res_all = (const float*)d_in[0];
    const float* res     = (const float*)d_in[1];
    const int*   starts  = (const int*)d_in[2];
    const int*   lens    = (const int*)d_in[3];
    const float* attn_W  = (const float*)d_in[4];
    const float* attn_b  = (const float*)d_in[5];
    const float* v_w     = (const float*)d_in[6];
    const float* id_W    = (const float*)d_in[7];
    const float* id_b    = (const float*)d_in[8];
    const float* slot_W  = (const float*)d_in[9];
    const float* slot_b  = (const float*)d_in[10];

    float* out_id = (float*)d_out;                   // [256,26]
    float* out_sf = out_id + BATCH * INTENT;         // [13056,122]

    char* ws = (char*)d_ws;                          // 1.4 MB scratch only
    unsigned short* Wb = (unsigned short*)ws;  ws += (size_t)HDIM * HDIM * 2;
    unsigned short* Sb = (unsigned short*)ws;

    pack_weights<<<(HDIM * HDIM + 255) / 256, 256, 0, stream>>>(attn_W, slot_W, Wb, Sb);
    fused_attn_slot<<<NWORD / 16, 256, 0, stream>>>(res_all, starts, lens, Wb, Sb,
                                                    attn_b, v_w, slot_b, out_sf);
    intent_fc<<<(BATCH * INTENT + 255) / 256, 256, 0, stream>>>(res, id_W, id_b, out_id);
}
